// Attention_34617436405903
// MI455X (gfx1250) — matrix operand order repaired and checked by reading, not compiled, not run on hardware
//
#include <hip/hip_runtime.h>

// ---------------------------------------------------------------------------
// MI455X (gfx1250) bipartite cross-attention, bf16 WMMA pipeline.
// GEMMs: v_wmma_f32_16x16x32_bf16 (wave32, f32 accum).
// Main GEMM stages tiles with the Tensor Data Mover (tensor_load_to_lds,
// TENSORcnt-tracked, double-buffered LDS).
// ---------------------------------------------------------------------------

#define C_DIM 1024
#define NHEAD 16
#define HD    64
#define BQ    128
#define BK    128
#define NK    256
#define MK    (BK * NK)   // 32768 rows of kx

typedef unsigned short ushort_t;
typedef __attribute__((ext_vector_type(8)))  float        v8f;
typedef __attribute__((ext_vector_type(8)))  __bf16       v8bf;
typedef __attribute__((ext_vector_type(16))) __bf16       v16bf;
typedef __attribute__((ext_vector_type(4)))  unsigned int u32x4;
typedef __attribute__((ext_vector_type(8)))  int          i32x8;
typedef __attribute__((ext_vector_type(4)))  int          i32x4;

__device__ __forceinline__ ushort_t f2bf(float f) {
    unsigned int u = __float_as_uint(f);
    u += 0x7FFFu + ((u >> 16) & 1u);   // round-to-nearest-even
    return (ushort_t)(u >> 16);
}

__device__ __forceinline__ v8f wmma_bf16(v16bf a, v16bf b, v8f c) {
    return __builtin_amdgcn_wmma_f32_16x16x32_bf16(false, a, false, b,
                                                   (short)0, c, false, false);
}

// A fragment (16x32 bf16, M x K): lane holds row m = lane&15.
// halves [0..7] = K kb8+0..7, halves [8..15] = K kb8+16..23, kb8 = 8*(lane>>4).
__device__ __forceinline__ v16bf load_a_frag(const ushort_t* tile, int stride, int lane) {
    const ushort_t* p = tile + (size_t)(lane & 15) * stride + ((lane >> 4) << 3);
    v8bf lo = *(const v8bf*)(p);
    v8bf hi = *(const v8bf*)(p + 16);
    return __builtin_shufflevector(lo, hi, 0,1,2,3,4,5,6,7,8,9,10,11,12,13,14,15);
}

// B fragment (32x16 bf16, K x N) sourced from B^T rows (row n contiguous in K):
// lane holds column n = lane&15, halves [0..7] = K kb8+0..7, [8..15] = K kb8+16..23, kb8 = 8*(lane>>4).
__device__ __forceinline__ v16bf load_b_frag(const ushort_t* tile, int stride, int lane) {
    const ushort_t* p = tile + (size_t)(lane & 15) * stride + ((lane >> 4) << 3);
    v8bf lo = *(const v8bf*)(p);
    v8bf hi = *(const v8bf*)(p + 16);
    return __builtin_shufflevector(lo, hi, 0,1,2,3,4,5,6,7,8,9,10,11,12,13,14,15);
}

// ---------------------------------------------------------------------------
// TDM: DMA a [rows x row_elems] bf16 tile (row stride gstride elems) from
// global into LDS at lds_off, padding +8 DWORDs (16 halves) after each row of
// row_elems halves so the LDS row stride is row_elems+16 halves.
// D# packing per CDNA5 ISA.  Issued per-wave; caller
// must guard so only one wave issues, then s_wait_tensorcnt + barrier.
// ---------------------------------------------------------------------------
__device__ __forceinline__ void tdm_load_tile(const ushort_t* gsrc, unsigned lds_off,
                                              int rows, int row_elems, int gstride,
                                              int pad_interval_code) {
    unsigned long long ga = (unsigned long long)(size_t)gsrc;
    u32x4 g0;
    g0[0] = 1u;                                            // count=1, user mode
    g0[1] = lds_off;                                       // lds_addr
    g0[2] = (unsigned)(ga & 0xFFFFFFFFull);                // global_addr[31:0]
    g0[3] = (unsigned)((ga >> 32) & 0x01FFFFFFull) | (2u << 30);  // addr[56:32] | type=2
    i32x8 g1;
    g1[0] = (1 << 16)            // data_size = 2 bytes
          | (1 << 20)            // pad_enable
          | (pad_interval_code << 22)
          | (7 << 25);           // pad_amount code 7 -> 8 DWORDs (16 halves)
    g1[1] = (row_elems & 0xFFFF) << 16;                        // tensor_dim0[15:0]
    g1[2] = ((unsigned)row_elems >> 16) | ((rows & 0xFFFF) << 16); // dim0[31:16] | dim1[15:0]
    g1[3] = ((unsigned)rows >> 16) | ((row_elems & 0xFFFF) << 16); // dim1[31:16] | tile_dim0
    g1[4] = rows & 0xFFFF;                                     // tile_dim1 | tile_dim2=0
    g1[5] = gstride;                                           // tensor_dim0_stride[31:0]
    g1[6] = 0;                                                 // stride0 hi | stride1 lo
    g1[7] = 0;                                                 // stride1 hi
    i32x4 z4 = {0, 0, 0, 0};
#if defined(__clang_major__) && (__clang_major__ >= 23)
    i32x8 z8 = {0, 0, 0, 0, 0, 0, 0, 0};
    __builtin_amdgcn_tensor_load_to_lds(g0, g1, z4, z4, z8, 0);
#else
    __builtin_amdgcn_tensor_load_to_lds(g0, g1, z4, z4, 0);
#endif
}

// ---------------------------------------------------------------------------
// K0: cast Wq / Wk (1024x1024 f32 each) to bf16.
// ---------------------------------------------------------------------------
__global__ __launch_bounds__(256)
void cast_weights(const float* __restrict__ Wq, const float* __restrict__ Wk,
                  ushort_t* __restrict__ wq_bf, ushort_t* __restrict__ wk_bf) {
    size_t i = (size_t)blockIdx.x * blockDim.x + threadIdx.x;
    float4 a = ((const float4*)Wq)[i];
    float4 b = ((const float4*)Wk)[i];
    ushort4 ua = { f2bf(a.x), f2bf(a.y), f2bf(a.z), f2bf(a.w) };
    ushort4 ub = { f2bf(b.x), f2bf(b.y), f2bf(b.z), f2bf(b.w) };
    ((ushort4*)wq_bf)[i] = ua;
    ((ushort4*)wk_bf)[i] = ub;
}

// ---------------------------------------------------------------------------
// K1: fused LayerNorm + bf16 cast for qx (128 rows) and kx (32768 rows).
// ---------------------------------------------------------------------------
__global__ __launch_bounds__(256)
void ln_cast(const float* __restrict__ qx, const float* __restrict__ kx,
             const float* __restrict__ gq, const float* __restrict__ bq,
             const float* __restrict__ gk, const float* __restrict__ bk,
             ushort_t* __restrict__ lnq, ushort_t* __restrict__ lnk) {
    const int row = blockIdx.x;
    const float* src; const float* g; const float* b; ushort_t* dst;
    if (row < MK) { src = kx + (size_t)row * C_DIM; dst = lnk + (size_t)row * C_DIM; g = gk; b = bk; }
    else { int r = row - MK; src = qx + (size_t)r * C_DIM; dst = lnq + (size_t)r * C_DIM; g = gq; b = bq; }

    const int t = threadIdx.x, wave = t >> 5, lane = t & 31;
    float4 v = ((const float4*)src)[t];
    float s  = v.x + v.y + v.z + v.w;
    float sq = v.x * v.x + v.y * v.y + v.z * v.z + v.w * v.w;
    #pragma unroll
    for (int off = 16; off >= 1; off >>= 1) {
        s  += __shfl_xor(s,  off, 32);
        sq += __shfl_xor(sq, off, 32);
    }
    __shared__ float ps[8], pq[8];
    if (lane == 0) { ps[wave] = s; pq[wave] = sq; }
    __syncthreads();
    float S = 0.f, Q = 0.f;
    #pragma unroll
    for (int i = 0; i < 8; ++i) { S += ps[i]; Q += pq[i]; }
    const float mean = S * (1.0f / C_DIM);
    const float var  = Q * (1.0f / C_DIM) - mean * mean;
    const float rstd = rsqrtf(var + 1e-5f);

    const int c0 = t * 4;
    ushort4 o;
    o.x = f2bf((v.x - mean) * rstd * g[c0 + 0] + b[c0 + 0]);
    o.y = f2bf((v.y - mean) * rstd * g[c0 + 1] + b[c0 + 1]);
    o.z = f2bf((v.z - mean) * rstd * g[c0 + 2] + b[c0 + 2]);
    o.w = f2bf((v.w - mean) * rstd * g[c0 + 3] + b[c0 + 3]);
    ((ushort4*)dst)[t] = o;
}

// ---------------------------------------------------------------------------
// K2: D[M][1024] = A[M][1024] @ W^T, bf16 in/out, f32 accum.
// 128x128 tile per 256-thread block, K-step 64, TDM-staged double-buffered
// LDS (row stride 80 halves via TDM pad).  Wave 0 drives the DMA.
// ---------------------------------------------------------------------------
#define KSTEP 64
#define GSTR  80    // KSTEP + 16 pad halves
__global__ __launch_bounds__(256)
void gemm_bf16(const ushort_t* __restrict__ A, const ushort_t* __restrict__ W,
               ushort_t* __restrict__ D) {
    const int m0 = blockIdx.x * 128, n0 = blockIdx.y * 128;
    const int tid = threadIdx.x, wave = tid >> 5, lane = tid & 31;
    __shared__ ushort_t sA[2][128 * GSTR];
    __shared__ ushort_t sB[2][128 * GSTR];

    v8f acc[8];
    v8f z = {};
    #pragma unroll
    for (int j = 0; j < 8; ++j) acc[j] = z;

    const ushort_t* Ab = A + (size_t)m0 * C_DIM;
    const ushort_t* Wb = W + (size_t)n0 * C_DIM;
    const int NSTEP = C_DIM / KSTEP;   // 16

    if (wave == 0) {   // waves 1-7 take the s_cbranch_execz path: no TDM issue
        tdm_load_tile(Ab, (unsigned)(size_t)&sA[0][0], 128, KSTEP, C_DIM, 4);
        tdm_load_tile(Wb, (unsigned)(size_t)&sB[0][0], 128, KSTEP, C_DIM, 4);
    }

    for (int s = 0; s < NSTEP; ++s) {
        const int buf = s & 1;
        if (wave == 0) {
            if (s + 1 < NSTEP) {   // prefetch next tile into other buffer
                tdm_load_tile(Ab + (s + 1) * KSTEP, (unsigned)(size_t)&sA[buf ^ 1][0],
                              128, KSTEP, C_DIM, 4);
                tdm_load_tile(Wb + (s + 1) * KSTEP, (unsigned)(size_t)&sB[buf ^ 1][0],
                              128, KSTEP, C_DIM, 4);
                __builtin_amdgcn_s_wait_tensorcnt(2);  // in-order: tile s has landed
            } else {
                __builtin_amdgcn_s_wait_tensorcnt(0);
            }
        }
        __syncthreads();
        #pragma unroll
        for (int kc = 0; kc < 2; ++kc) {
            v16bf af = load_a_frag(&sA[buf][(wave * 16) * GSTR + kc * 32], GSTR, lane);
            v16bf bfr[8];
            #pragma unroll
            for (int nt = 0; nt < 8; ++nt)
                bfr[nt] = load_b_frag(&sB[buf][(nt * 16) * GSTR + kc * 32], GSTR, lane);
            #pragma unroll
            for (int nt = 0; nt < 8; ++nt)
                acc[nt] = wmma_bf16(af, bfr[nt], acc[nt]);
        }
        __syncthreads();
    }
    const int nl = lane & 15, mh = (lane >> 4) * 8;
    #pragma unroll
    for (int nt = 0; nt < 8; ++nt)
        #pragma unroll
        for (int j = 0; j < 8; ++j)
            D[(size_t)(m0 + wave * 16 + mh + j) * C_DIM + n0 + nt * 16 + nl] = f2bf(acc[nt][j]);
}

// ---------------------------------------------------------------------------
// K3: attention scores + softmax(axis=Nk) + head-sum -> w[q][kb][n] (bf16).
// ---------------------------------------------------------------------------
__global__ __launch_bounds__(256)
void attn_w(const ushort_t* __restrict__ qp, const ushort_t* __restrict__ kp,
            ushort_t* __restrict__ w_bf) {
    const int kb = blockIdx.x, q0 = blockIdx.y * 32;
    const int tid = threadIdx.x, wave = tid >> 5, lane = tid & 31;
    __shared__ float sS[32][256];
    __shared__ float sW[32][256];
    for (int i = tid; i < 32 * 256; i += 256) ((float*)sW)[i] = 0.f;
    const float scale = 0.125f;   // 64^-0.5

    for (int h = 0; h < NHEAD; ++h) {
        __syncthreads();
        #pragma unroll
        for (int mt = 0; mt < 2; ++mt) {
            #pragma unroll
            for (int nt2 = 0; nt2 < 2; ++nt2) {
                const int nt = wave * 2 + nt2;
                v8f acc = {};
                #pragma unroll
                for (int ks = 0; ks < 2; ++ks) {
                    v16bf a = load_a_frag(qp + (size_t)(q0 + mt * 16) * C_DIM + h * HD + ks * 32,
                                          C_DIM, lane);
                    v16bf b = load_b_frag(kp + (size_t)(kb * NK + nt * 16) * C_DIM + h * HD + ks * 32,
                                          C_DIM, lane);
                    acc = wmma_bf16(a, b, acc);
                }
                const int n = nt * 16 + (lane & 15), mh = (lane >> 4) * 8;
                #pragma unroll
                for (int j = 0; j < 8; ++j) sS[mt * 16 + mh + j][n] = acc[j] * scale;
            }
        }
        __syncthreads();
        #pragma unroll
        for (int rr = 0; rr < 4; ++rr) {
            const int r = wave * 4 + rr;
            float vals[8];
            float vmax = -1e30f;
            #pragma unroll
            for (int j = 0; j < 8; ++j) { vals[j] = sS[r][lane * 8 + j]; vmax = fmaxf(vmax, vals[j]); }
            #pragma unroll
            for (int off = 16; off >= 1; off >>= 1) vmax = fmaxf(vmax, __shfl_xor(vmax, off, 32));
            float ssum = 0.f;
            #pragma unroll
            for (int j = 0; j < 8; ++j) { vals[j] = __expf(vals[j] - vmax); ssum += vals[j]; }
            #pragma unroll
            for (int off = 16; off >= 1; off >>= 1) ssum += __shfl_xor(ssum, off, 32);
            const float rinv = __frcp_rn(ssum);
            #pragma unroll
            for (int j = 0; j < 8; ++j) sW[r][lane * 8 + j] += vals[j] * rinv;
        }
    }
    __syncthreads();
    for (int i = tid; i < 32 * 32; i += 256) {
        const int r = i >> 5, ch = (i & 31) * 8;
        const size_t base = ((size_t)(q0 + r) * BK + kb) * NK + ch;
        #pragma unroll
        for (int j = 0; j < 8; ++j) w_bf[base + j] = f2bf(sW[r][ch + j]);
    }
}

// ---------------------------------------------------------------------------
// K4: out[q][kb][c] = sum_n w[q][kb][n] * kx[kb][n][c]   (f32 output)
// B = kx tile cast + transposed through LDS (per-lane contiguous-K columns).
// ---------------------------------------------------------------------------
#define LSTR 48
__global__ __launch_bounds__(256)
void out_gemm(const ushort_t* __restrict__ w_bf, const float* __restrict__ kx,
              float* __restrict__ out) {
    const int kb = blockIdx.x, c0 = blockIdx.y * 128;
    const int tid = threadIdx.x, wave = tid >> 5, lane = tid & 31;
    __shared__ ushort_t sB[128 * LSTR];

    v8f acc[8];
    v8f z = {};
    #pragma unroll
    for (int j = 0; j < 8; ++j) acc[j] = z;

    const int wrow_stride = BK * NK;   // 32768 halves between q rows
    for (int k0 = 0; k0 < NK; k0 += 32) {
        __syncthreads();
        #pragma unroll
        for (int it = 0; it < 4; ++it) {
            int idx = tid + it * 256;
            int n = idx >> 5, cc = (idx & 31) * 4;
            float4 v = *(const float4*)&kx[((size_t)kb * NK + k0 + n) * C_DIM + c0 + cc];
            sB[(cc + 0) * LSTR + n] = f2bf(v.x);
            sB[(cc + 1) * LSTR + n] = f2bf(v.y);
            sB[(cc + 2) * LSTR + n] = f2bf(v.z);
            sB[(cc + 3) * LSTR + n] = f2bf(v.w);
        }
        __syncthreads();
        v16bf a = load_a_frag(w_bf + (size_t)(wave * 16) * wrow_stride + (size_t)kb * NK + k0,
                              wrow_stride, lane);
        v16bf bfr[8];
        #pragma unroll
        for (int nt = 0; nt < 8; ++nt)
            bfr[nt] = load_b_frag(&sB[(nt * 16) * LSTR], LSTR, lane);
        #pragma unroll
        for (int nt = 0; nt < 8; ++nt)
            acc[nt] = wmma_bf16(a, bfr[nt], acc[nt]);
    }
    const int nl = lane & 15, mh = (lane >> 4) * 8;
    #pragma unroll
    for (int nt = 0; nt < 8; ++nt)
        #pragma unroll
        for (int j = 0; j < 8; ++j)
            out[((size_t)(wave * 16 + mh + j) * BK + kb) * C_DIM + c0 + nt * 16 + nl] = acc[nt][j];
}

// ---------------------------------------------------------------------------
extern "C" void kernel_launch(void* const* d_in, const int* in_sizes, int n_in,
                              void* d_out, int out_size, void* d_ws, size_t ws_size,
                              hipStream_t stream) {
    const float* qx = (const float*)d_in[0];
    const float* kx = (const float*)d_in[1];
    const float* gq = (const float*)d_in[2];
    const float* bq = (const float*)d_in[3];
    const float* gk = (const float*)d_in[4];
    const float* bk = (const float*)d_in[5];
    const float* Wq = (const float*)d_in[6];
    const float* Wk = (const float*)d_in[7];
    float* out = (float*)d_out;

    char* ws = (char*)d_ws;
    const size_t MB = 1ull << 20;
    ushort_t* wq_bf  = (ushort_t*)(ws + 0 * MB);    //  2 MB
    ushort_t* wk_bf  = (ushort_t*)(ws + 2 * MB);    //  2 MB
    ushort_t* lnq_bf = (ushort_t*)(ws + 4 * MB);    //  0.25 MB
    ushort_t* qp_bf  = (ushort_t*)(ws + 6 * MB);    //  0.25 MB
    ushort_t* lnk_bf = (ushort_t*)(ws + 8 * MB);    // 64 MB
    ushort_t* kp_bf  = (ushort_t*)(ws + 72 * MB);   // 64 MB
    ushort_t* w_bf   = (ushort_t*)(ws + 136 * MB);  //  8 MB

    cast_weights<<<dim3((C_DIM * C_DIM) / (4 * 256)), dim3(256), 0, stream>>>(Wq, Wk, wq_bf, wk_bf);
    ln_cast<<<dim3(MK + BQ), dim3(256), 0, stream>>>(qx, kx, gq, bq, gk, bk, lnq_bf, lnk_bf);
    gemm_bf16<<<dim3(MK / 128, C_DIM / 128), dim3(256), 0, stream>>>(lnk_bf, wk_bf, kp_bf);
    gemm_bf16<<<dim3(BQ / 128, C_DIM / 128), dim3(256), 0, stream>>>(lnq_bf, wq_bf, qp_bf);
    attn_w<<<dim3(BK, BQ / 32), dim3(256), 0, stream>>>(qp_bf, kp_bf, w_bf);
    out_gemm<<<dim3(BK, C_DIM / 128), dim3(256), 0, stream>>>(w_bf, kx, out);
}
